// MultiHeadAttention_50208167690550
// MI455X (gfx1250) — matrix operand order repaired and checked by reading, not compiled, not run on hardware
//
#include <hip/hip_runtime.h>

// MI455X / gfx1250 fused multi-head attention.
// B=2, S=2048, H=768, NH=12, D=64. fp32 in/out, bf16 WMMA internally
// (v_wmma_f32_16x16x32_bf16, f32 accumulate).
//
// Workspace (all __bf16):
//   Xq,Xk,Xv : [B*S, H]   converted inputs     (3 x 4096*768)
//   Wqb,Wkb,Wvb : [H, H]  converted weights    (3 x 768*768)
//   Qbf,Kbf  : [B*S, H]   projected, row-major
//   Vt       : [B, NH, D, S] projected, per-head transposed
// total ~41.3 MB

#define Bc   2
#define Sc   2048
#define Hc   768
#define NHc  12
#define Dc   64
#define Mc   (Bc * Sc)   // 4096

#define NWAVES 8
#define KBW    (Sc / NWAVES)   // 256 keys per wave

typedef __bf16 bf16_t;
typedef __attribute__((ext_vector_type(16))) __bf16 v16bf;
typedef __attribute__((ext_vector_type(8)))  __bf16 v8bf;
typedef __attribute__((ext_vector_type(4)))  __bf16 v4bf;
typedef __attribute__((ext_vector_type(8)))  float  v8f;

union frag16 {
  v16bf v;
  struct { v8bf lo, hi; } s;
};

// ---------------------------------------------------------------------------
// f32 -> bf16 elementwise convert (bandwidth bound), 4 elems/thread
// ---------------------------------------------------------------------------
__global__ void __launch_bounds__(256)
cvt_kernel(const float* __restrict__ src, bf16_t* __restrict__ dst, int n4) {
  int i = blockIdx.x * 256 + threadIdx.x;
  if (i < n4) {
    float4 v = ((const float4*)src)[i];
    v4bf o;
    o[0] = (__bf16)v.x; o[1] = (__bf16)v.y;
    o[2] = (__bf16)v.z; o[3] = (__bf16)v.w;
    ((v4bf*)dst)[i] = o;
  }
}

// ---------------------------------------------------------------------------
// QKV projection (bf16 in, bf16 out): Y = X @ W^T + bias
// One wave computes a 16x64 tile.  mode 0 -> out[m*Hc + n] row-major
// mode 1 -> Vt[((b*NH + h)*D + d)*S + s]  (per-head transposed)
// ---------------------------------------------------------------------------
__global__ void __launch_bounds__(32)
qkv_proj_kernel(const bf16_t* __restrict__ X, const bf16_t* __restrict__ W,
                const float* __restrict__ bias, bf16_t* __restrict__ out,
                int mode) {
  const int lane   = threadIdx.x;
  const int m0     = blockIdx.x * 16;
  const int n0     = blockIdx.y * 64;
  const int lm     = lane & 15;
  const int hiHalf = (lane >= 16) ? 1 : 0;

  const bf16_t* xrow = X + (size_t)(m0 + lm) * Hc;
  const int ad0 = hiHalf ? 16 : 0;   // A K-run start
  const int bk0 = hiHalf ? 16 : 0;   // B K-run start

  v8f zero = {};
  v8f acc[4];
#pragma unroll
  for (int t = 0; t < 4; ++t) acc[t] = zero;

  for (int k0 = 0; k0 < Hc; k0 += 32) {
    frag16 a;
    a.s.lo = *(const v8bf*)(xrow + k0 + ad0);        // K = ad0 .. ad0+7
    a.s.hi = *(const v8bf*)(xrow + k0 + ad0 + 8);    // K = ad0+8 ..
#pragma unroll
    for (int t = 0; t < 4; ++t) {
      const int n = n0 + t * 16 + lm;
      v16bf b = *(const v16bf*)(W + (size_t)n * Hc + k0 + bk0);
      acc[t] = __builtin_amdgcn_wmma_f32_16x16x32_bf16(
          false, a.v, false, b, (short)0, acc[t], false, false);
    }
  }

  // epilogue: + bias, cvt bf16, store
#pragma unroll
  for (int t = 0; t < 4; ++t) {
    const int   n  = n0 + t * 16 + lm;
    const float bn = bias[n];
    if (mode == 0) {
#pragma unroll
      for (int r = 0; r < 8; ++r) {
        const int m = m0 + r + (hiHalf ? 8 : 0);
        out[(size_t)m * Hc + n] = (__bf16)(acc[t][r] + bn);
      }
    } else {
      // 8 consecutive s values for this lane -> one 16B store
      v8bf o;
#pragma unroll
      for (int r = 0; r < 8; ++r) o[r] = (__bf16)(acc[t][r] + bn);
      const int bb = m0 >> 11;                    // batch
      const int s0 = (m0 & (Sc - 1)) + (hiHalf ? 8 : 0);
      const int h  = n >> 6;
      const int d  = n & (Dc - 1);
      *(v8bf*)(out + ((((size_t)bb * NHc + h) * Dc) + d) * Sc + s0) = o;
    }
  }
}

// ---------------------------------------------------------------------------
// Attention: one workgroup (8 waves, 256 threads) per (b, h, 16-query tile).
// Waves partition the key dimension (256 keys each) so all 4 SIMD32s of the
// WGP run WMMA with 2 waves each for latency hiding.
// LDS: f32 scores 128KB (reused for O partials) + bf16 probs 64KB + mask 8KB.
// ---------------------------------------------------------------------------
__global__ void __launch_bounds__(256)
attention_kernel(const bf16_t* __restrict__ Qbf, const bf16_t* __restrict__ Kbf,
                 const bf16_t* __restrict__ Vt, const float* __restrict__ mask,
                 float* __restrict__ out) {
  extern __shared__ float smem[];
  float*  sS    = smem;                              // 16*Sc f32 scores / O partials
  bf16_t* sP    = (bf16_t*)(smem + 16 * Sc);         // 16*Sc bf16 probs
  float*  sMask = (float*)(sP + 16 * Sc);            // Sc
  float*  sRsum = sMask + Sc;                        // 16

  const int tid    = threadIdx.x;
  const int w      = tid >> 5;        // wave id 0..7
  const int lane   = tid & 31;
  const int q0     = blockIdx.x * 16;
  const int h      = blockIdx.y;
  const int b      = blockIdx.z;
  const int lm     = lane & 15;
  const int hiHalf = (lane >= 16) ? 1 : 0;
  const int ad0    = hiHalf ? 16 : 0;
  const int bk0    = hiHalf ? 16 : 0;

  // additive mask -> LDS (cooperative)
  const float* mrow = mask + (size_t)b * Sc;
  for (int i = tid; i < Sc; i += 256) sMask[i] = mrow[i];

  // Q A-fragments (d = 0..31 and 32..63); same tile in every wave (L2 hits)
  const bf16_t* qptr = Qbf + ((size_t)(b * Sc + q0 + lm) * Hc + h * Dc);
  frag16 aQ0, aQ1;
  aQ0.s.lo = *(const v8bf*)(qptr + ad0);
  aQ0.s.hi = *(const v8bf*)(qptr + ad0 + 8);
  aQ1.s.lo = *(const v8bf*)(qptr + 32 + ad0);
  aQ1.s.hi = *(const v8bf*)(qptr + 32 + ad0 + 8);

  __syncthreads();

  // ---- pass 1: scores = (Q K^T) * 1/sqrt(D) + mask -> LDS (per-wave slice) --
  const float scale = 0.125f;  // 1/sqrt(64)
  const int klo = w * KBW, khi = klo + KBW;
  for (int kb = klo; kb < khi; kb += 16) {
    const bf16_t* kptr = Kbf + ((size_t)(b * Sc + kb + lm) * Hc + h * Dc);
    v16bf b0 = *(const v16bf*)(kptr + bk0);        // d = bk0 .. bk0+15
    v16bf b1 = *(const v16bf*)(kptr + 32 + bk0);   // d = 32+bk0 ..
    v8f c = {};
    c = __builtin_amdgcn_wmma_f32_16x16x32_bf16(false, aQ0.v, false, b0,
                                                (short)0, c, false, false);
    c = __builtin_amdgcn_wmma_f32_16x16x32_bf16(false, aQ1.v, false, b1,
                                                (short)0, c, false, false);
    if (kb + 16 < khi)  // stream hint for next K tile
      __builtin_prefetch(Kbf + ((size_t)(b * Sc + kb + 16 + lm) * Hc + h * Dc),
                         0, 1);
    const int key = kb + lm;
    const float mk = sMask[key];
#pragma unroll
    for (int r = 0; r < 8; ++r) {
      const int m = r + (hiHalf ? 8 : 0);
      sS[m * Sc + key] = c[r] * scale + mk;
    }
  }

  __syncthreads();

  // ---- softmax over keys: 16 lanes per row (row stays inside one wave);
  //      exp -> bf16 probs in sP; 1/rowsum folded into the output scale ----
  {
    const int row = tid >> 4;                 // 0..15
    const int off = (tid & 15) * (Sc / 16);   // 128-wide segment
    const float* srow = sS + row * Sc + off;
    bf16_t*      prow = sP + row * Sc + off;
    float mx = -3.4e38f;
    for (int j = 0; j < Sc / 16; j += 4) {
      float4 v = *(const float4*)(srow + j);
      mx = fmaxf(mx, fmaxf(fmaxf(v.x, v.y), fmaxf(v.z, v.w)));
    }
    mx = fmaxf(mx, __shfl_xor(mx, 1, 32));
    mx = fmaxf(mx, __shfl_xor(mx, 2, 32));
    mx = fmaxf(mx, __shfl_xor(mx, 4, 32));
    mx = fmaxf(mx, __shfl_xor(mx, 8, 32));
    float sum = 0.f;
    for (int j = 0; j < Sc / 16; j += 4) {
      float4 v = *(const float4*)(srow + j);
      float e0 = __expf(v.x - mx), e1 = __expf(v.y - mx);
      float e2 = __expf(v.z - mx), e3 = __expf(v.w - mx);
      v4bf p;
      p[0] = (__bf16)e0; p[1] = (__bf16)e1;
      p[2] = (__bf16)e2; p[3] = (__bf16)e3;
      *(v4bf*)(prow + j) = p;
      sum += (e0 + e1) + (e2 + e3);
    }
    sum += __shfl_xor(sum, 1, 32);
    sum += __shfl_xor(sum, 2, 32);
    sum += __shfl_xor(sum, 4, 32);
    sum += __shfl_xor(sum, 8, 32);
    if ((tid & 15) == 0) sRsum[row] = sum;
  }

  __syncthreads();

  // ---- pass 2: partial O = P~ V over this wave's key slice ----
  v8f zero = {};
  v8f acc[4];
#pragma unroll
  for (int t = 0; t < 4; ++t) acc[t] = zero;

  const size_t vbase = ((size_t)(b * NHc + h)) * Dc * Sc;
  for (int kb = klo; kb < khi; kb += 32) {
    const bf16_t* prow = sP + lm * Sc + kb + ad0;
    frag16 aP;
    aP.s.lo = *(const v8bf*)(prow);
    aP.s.hi = *(const v8bf*)(prow + 8);
#pragma unroll
    for (int t = 0; t < 4; ++t) {
      const int d = t * 16 + lm;
      v16bf bV = *(const v16bf*)(Vt + vbase + (size_t)d * Sc + kb + bk0);
      acc[t] = __builtin_amdgcn_wmma_f32_16x16x32_bf16(
          false, aP.v, false, bV, (short)0, acc[t], false, false);
    }
  }

  // spill partial 16x64 O into the (dead) score region: wave w at sS+w*1024
  {
    float* sOp = sS + w * (16 * Dc);
#pragma unroll
    for (int t = 0; t < 4; ++t) {
#pragma unroll
      for (int r = 0; r < 8; ++r) {
        const int m = r + (hiHalf ? 8 : 0);
        sOp[m * Dc + t * 16 + lm] = acc[t][r];
      }
    }
  }

  __syncthreads();

  // ---- final: reduce 8 partials, scale by 1/rowsum, b128 store ----
  {
    const int m  = tid >> 4;          // 0..15
    const int d0 = (tid & 15) * 4;    // 0..60
    float4 s = make_float4(0.f, 0.f, 0.f, 0.f);
#pragma unroll
    for (int ww = 0; ww < NWAVES; ++ww) {
      float4 p = *(const float4*)(sS + ww * (16 * Dc) + m * Dc + d0);
      s.x += p.x; s.y += p.y; s.z += p.z; s.w += p.w;
    }
    const float inv = 1.0f / sRsum[m];
    s.x *= inv; s.y *= inv; s.z *= inv; s.w *= inv;
    *(float4*)(out + (size_t)(b * Sc + q0 + m) * Hc + h * Dc + d0) = s;
  }
}

// ---------------------------------------------------------------------------
extern "C" void kernel_launch(void* const* d_in, const int* in_sizes, int n_in,
                              void* d_out, int out_size, void* d_ws,
                              size_t ws_size, hipStream_t stream) {
  const float* query = (const float*)d_in[0];
  const float* key   = (const float*)d_in[1];
  const float* value = (const float*)d_in[2];
  const float* amask = (const float*)d_in[3];
  const float* Wq = (const float*)d_in[4];
  const float* bq = (const float*)d_in[5];
  const float* Wk = (const float*)d_in[6];
  const float* bk = (const float*)d_in[7];
  const float* Wv = (const float*)d_in[8];
  const float* bv = (const float*)d_in[9];
  float* out = (float*)d_out;

  const size_t nX = (size_t)Mc * Hc;   // 3,145,728
  const size_t nW = (size_t)Hc * Hc;   //   589,824

  bf16_t* Xq  = (bf16_t*)d_ws;
  bf16_t* Xk  = Xq + nX;
  bf16_t* Xv  = Xk + nX;
  bf16_t* Wqb = Xv + nX;
  bf16_t* Wkb = Wqb + nW;
  bf16_t* Wvb = Wkb + nW;
  bf16_t* Qbf = Wvb + nW;
  bf16_t* Kbf = Qbf + nX;
  bf16_t* Vt  = Kbf + nX;

  // 1) convert inputs + weights to bf16
  {
    const int nx4 = (int)(nX / 4), nw4 = (int)(nW / 4);
    dim3 gx((nx4 + 255) / 256), gw((nw4 + 255) / 256);
    cvt_kernel<<<gx, 256, 0, stream>>>(query, Xq, nx4);
    cvt_kernel<<<gx, 256, 0, stream>>>(key,   Xk, nx4);
    cvt_kernel<<<gx, 256, 0, stream>>>(value, Xv, nx4);
    cvt_kernel<<<gw, 256, 0, stream>>>(Wq, Wqb, nw4);
    cvt_kernel<<<gw, 256, 0, stream>>>(Wk, Wkb, nw4);
    cvt_kernel<<<gw, 256, 0, stream>>>(Wv, Wvb, nw4);
  }

  // 2) projections (WMMA GEMM, bf16 operands)
  dim3 pgrid(Mc / 16, Hc / 64);
  qkv_proj_kernel<<<pgrid, 32, 0, stream>>>(Xq, Wqb, bq, Qbf, 0);
  qkv_proj_kernel<<<pgrid, 32, 0, stream>>>(Xk, Wkb, bk, Kbf, 0);
  qkv_proj_kernel<<<pgrid, 32, 0, stream>>>(Xv, Wvb, bv, Vt,  1);

  // 3) attention
  dim3 agrid(Sc / 16, NHc, Bc);
  size_t smem = (size_t)(16 * Sc) * 4      // f32 scores / O partials
              + (size_t)(16 * Sc) * 2      // bf16 probs
              + (size_t)Sc * 4 + 16 * 4;   // mask + rowsums  (~205 KB)
  attention_kernel<<<agrid, 256, smem, stream>>>(Qbf, Kbf, Vt, amask, out);
}
